// PTBModel_78262894067862
// MI455X (gfx1250) — matrix operand order repaired and checked by reading, not compiled, not run on hardware
//
#include <hip/hip_runtime.h>
#include <stdint.h>

// Problem dimensions (fixed by the reference).
#define TDIM 256
#define BDIM 64
#define VDIM 10000
#define DDIM 256
#define HDIM 1024

typedef __attribute__((ext_vector_type(16))) __bf16 bf16x16;
typedef __attribute__((ext_vector_type(8)))  __bf16 bf16x8;
typedef __attribute__((ext_vector_type(8)))  float  f32x8;
typedef int v4i __attribute__((vector_size(16)));

union BFrag {
  bf16x16 v;
  bf16x8  h[2];
};

// Address-space qualified pointer types for the async-to-LDS builtin
// (diagnostics show param0 is AS(1) int4*; AS(3) assumed for the LDS side).
typedef __attribute__((address_space(1))) v4i* gbl_v4i_ptr;
typedef __attribute__((address_space(3))) v4i* lds_v4i_ptr;

// Async copy of 16 bytes global -> LDS (gfx1250 GLOBAL_LOAD_ASYNC_TO_LDS_B128).
__device__ __forceinline__ void async_cp16(void* lds_dst, const void* gsrc) {
#if __has_builtin(__builtin_amdgcn_global_load_async_to_lds_b128)
  __builtin_amdgcn_global_load_async_to_lds_b128(
      (gbl_v4i_ptr)(uintptr_t)gsrc,
      (lds_v4i_ptr)(uint32_t)(uintptr_t)lds_dst,
      /*offset=*/0, /*cpol=*/0);
#else
  *(v4i*)lds_dst = *(const v4i*)gsrc;
#endif
}

__device__ __forceinline__ void wait_async() {
#if __has_builtin(__builtin_amdgcn_global_load_async_to_lds_b128)
#if __has_builtin(__builtin_amdgcn_s_wait_asynccnt)
  __builtin_amdgcn_s_wait_asynccnt(0);
#else
  asm volatile("s_wait_asynccnt 0x0" ::: "memory");
#endif
#endif
}

// ---------------------------------------------------------------------------
// Fragment loaders (CDNA5 ISA 7.12.2 wave32 layouts).
//   A 16x32 bf16: lane l -> row (l&15); halves 0..7 = K k0+(l>>4)*8..+7,
//                 halves 8..15 = +16.    (two contiguous 16B loads)
//   B 32x16 bf16: lane l -> col (l&15); same K order as A:
//                 halves 0..7 = K k0+(l>>4)*8..+7, halves 8..15 = +16.
// ---------------------------------------------------------------------------
__device__ __forceinline__ bf16x16 load_a_frag(const __bf16* arow, int k0, int grp) {
  BFrag a;
  a.h[0] = *reinterpret_cast<const bf16x8*>(arow + k0 + grp * 8);
  a.h[1] = *reinterpret_cast<const bf16x8*>(arow + k0 + 16 + grp * 8);
  return a.v;
}
__device__ __forceinline__ bf16x16 load_b_frag(const __bf16* wrow, int k0, int grp) {
  BFrag b;
  b.h[0] = *reinterpret_cast<const bf16x8*>(wrow + k0 + grp * 8);
  b.h[1] = *reinterpret_cast<const bf16x8*>(wrow + k0 + 16 + grp * 8);
  return b.v;
}
__device__ __forceinline__ f32x8 wmma_bf16(bf16x16 a, bf16x16 b, f32x8 c) {
  return __builtin_amdgcn_wmma_f32_16x16x32_bf16(false, a, false, b, (short)0, c,
                                                 false, false);
}

// ---------------------------------------------------------------------------
// f32 -> bf16 cast (weight conversion)
// ---------------------------------------------------------------------------
__global__ __launch_bounds__(256) void cast_bf16_kernel(
    const float* __restrict__ src, __bf16* __restrict__ dst, int n)
{
  int i = blockIdx.x * blockDim.x + threadIdx.x;
  if (i < n) dst[i] = (__bf16)src[i];
}

// ---------------------------------------------------------------------------
// Embedding gather: X[t*B+b, d] = (bf16) emb[inputs[t,b], d]
// ---------------------------------------------------------------------------
__global__ __launch_bounds__(256) void embed_kernel(
    const int* __restrict__ ids, const float* __restrict__ emb,
    __bf16* __restrict__ X)
{
  int i = blockIdx.x * blockDim.x + threadIdx.x;
  const int n = TDIM * BDIM * DDIM;
  if (i >= n) return;
  int tb = i / DDIM;
  int d  = i - tb * DDIM;
  int tok = ids[tb];
  X[i] = (__bf16)emb[(size_t)tok * DDIM + d];
}

// ---------------------------------------------------------------------------
// Initialize running state: h/c f32 (aliased to d_out hT/cT) + h bf16.
// ---------------------------------------------------------------------------
__global__ __launch_bounds__(256) void init_state_kernel(
    const float* __restrict__ h0, const float* __restrict__ c0,
    float* __restrict__ h, float* __restrict__ c, __bf16* __restrict__ hbf)
{
  int i = blockIdx.x * blockDim.x + threadIdx.x;
  if (i >= BDIM * HDIM) return;
  float hv = h0[i];
  h[i]   = hv;
  c[i]   = c0[i];
  hbf[i] = (__bf16)hv;
}

// ---------------------------------------------------------------------------
// Per-step gates GEMM: gates[B,4H] = Xt*Wih^T + hprev*Whh^T + bias
// 32x32 register tile per wave (4 wmma / k-step): 2 x 128 = 256 tiles,
// 8 waves/block -> 32 blocks. K is fused: 256 (input) + 1024 (recurrent).
// ---------------------------------------------------------------------------
__global__ __launch_bounds__(256) void lstm_gates_kernel(
    const __bf16* __restrict__ Xt,    // [B, D] bf16
    const __bf16* __restrict__ hprev, // [B, H] bf16
    const __bf16* __restrict__ Wih,   // [4H, D] bf16
    const __bf16* __restrict__ Whh,   // [4H, H] bf16
    const float*  __restrict__ bias,  // [4H]
    float* __restrict__ gates)        // [B, 4H]
{
  const int NT = (4 * HDIM) / 32;               // 128 n-tiles of 32
  int tile = blockIdx.x * (blockDim.x >> 5) + (threadIdx.x >> 5);
  if (tile >= (BDIM / 32) * NT) return;         // wave-uniform
  int tm = tile / NT, tn = tile - tm * NT;
  const int wm = tm * 32, wn = tn * 32;

  const int lane = threadIdx.x & 31;
  const int grp = lane >> 4, l16 = lane & 15;

  f32x8 acc00 = {}, acc01 = {}, acc10 = {}, acc11 = {};

  // Input contribution (K = D).
  {
    const __bf16* a0 = Xt + (size_t)(wm + l16) * DDIM;
    const __bf16* a1 = Xt + (size_t)(wm + 16 + l16) * DDIM;
    const __bf16* w0 = Wih + (size_t)(wn + l16) * DDIM;
    const __bf16* w1 = Wih + (size_t)(wn + 16 + l16) * DDIM;
#pragma unroll 2
    for (int k0 = 0; k0 < DDIM; k0 += 32) {
      bf16x16 fa0 = load_a_frag(a0, k0, grp);
      bf16x16 fa1 = load_a_frag(a1, k0, grp);
      bf16x16 fb0 = load_b_frag(w0, k0, grp);
      bf16x16 fb1 = load_b_frag(w1, k0, grp);
      acc00 = wmma_bf16(fa0, fb0, acc00);
      acc01 = wmma_bf16(fa0, fb1, acc01);
      acc10 = wmma_bf16(fa1, fb0, acc10);
      acc11 = wmma_bf16(fa1, fb1, acc11);
    }
  }
  // Recurrent contribution (K = H).
  {
    const __bf16* a0 = hprev + (size_t)(wm + l16) * HDIM;
    const __bf16* a1 = hprev + (size_t)(wm + 16 + l16) * HDIM;
    const __bf16* w0 = Whh + (size_t)(wn + l16) * HDIM;
    const __bf16* w1 = Whh + (size_t)(wn + 16 + l16) * HDIM;
#pragma unroll 2
    for (int k0 = 0; k0 < HDIM; k0 += 32) {
      bf16x16 fa0 = load_a_frag(a0, k0, grp);
      bf16x16 fa1 = load_a_frag(a1, k0, grp);
      bf16x16 fb0 = load_b_frag(w0, k0, grp);
      bf16x16 fb1 = load_b_frag(w1, k0, grp);
      acc00 = wmma_bf16(fa0, fb0, acc00);
      acc01 = wmma_bf16(fa0, fb1, acc01);
      acc10 = wmma_bf16(fa1, fb0, acc10);
      acc11 = wmma_bf16(fa1, fb1, acc11);
    }
  }

  // Epilogue: + bias, store 32x32.
  float bn0 = bias[wn + l16];
  float bn1 = bias[wn + 16 + l16];
  const f32x8* accs[2][2] = {{&acc00, &acc01}, {&acc10, &acc11}};
#pragma unroll
  for (int mm = 0; mm < 2; ++mm) {
#pragma unroll
    for (int r = 0; r < 8; ++r) {
      size_t row = (size_t)(wm + mm * 16 + grp * 8 + r);
      gates[row * (4 * HDIM) + wn + l16]      = (*accs[mm][0])[r] + bn0;
      gates[row * (4 * HDIM) + wn + 16 + l16] = (*accs[mm][1])[r] + bn1;
    }
  }
}

// ---------------------------------------------------------------------------
// LSTM cell pointwise update.
// ---------------------------------------------------------------------------
__device__ __forceinline__ float sigf(float x) {
  return 1.0f / (1.0f + __expf(-x));
}

__global__ __launch_bounds__(256) void lstm_cell_kernel(
    const float* __restrict__ gates, // [B, 4H]
    float* __restrict__ h,           // [B, H] running f32 (d_out hT)
    float* __restrict__ c,           // [B, H] running f32 (d_out cT)
    __bf16* __restrict__ h_bf)       // [B, H] -> hs_bf16 slot t
{
  int i = blockIdx.x * blockDim.x + threadIdx.x;
  if (i >= BDIM * HDIM) return;
  int b  = i / HDIM;
  int hh = i - b * HDIM;
  const float* g = gates + (size_t)b * (4 * HDIM);
  float gi = g[hh];
  float gf = g[HDIM + hh];
  float gg = g[2 * HDIM + hh];
  float go = g[3 * HDIM + hh];
  float cv = sigf(gf) * c[i] + sigf(gi) * tanhf(gg);
  float hv = sigf(go) * tanhf(cv);
  c[i]    = cv;
  h[i]    = hv;
  h_bf[i] = (__bf16)hv;
}

// ---------------------------------------------------------------------------
// Output projection: logits[TB, V] = hs*Wout^T + b_out
// Block tile 256(M) x 32(N); 8 waves, each a 32x32 register tile.
// The 32-column weight strip is shared by all 8 waves and staged through LDS
// with async global->LDS copies, in two K=512 chunks.
// LDS rows padded (+8 halves) so 16-lane fragment reads are conflict-free.
// ---------------------------------------------------------------------------
#define PROJ_BM 256
#define PROJ_BN 32
#define KCHUNK  512
#define LDSROW  (KCHUNK + 8)   // halves; 1040B row stride -> bank stride 4
#define NBLK    ((VDIM + PROJ_BN - 1) / PROJ_BN)  // 313 (last block 16 cols)

__global__ __launch_bounds__(256) void proj_kernel(
    const __bf16* __restrict__ hs,   // [TB, H] bf16
    const __bf16* __restrict__ Wout, // [V, H] bf16
    const float*  __restrict__ bout, // [V]
    float* __restrict__ logits)      // [TB, V]
{
  __shared__ __bf16 lds_w[PROJ_BN * LDSROW];    // 33,280 B

  const int mb = blockIdx.x / NBLK;
  const int nb = blockIdx.x - mb * NBLK;
  const int m0 = mb * PROJ_BM;
  const int n0 = nb * PROJ_BN;

  const int tid  = threadIdx.x;
  const int wave = tid >> 5;
  const int lane = tid & 31;
  const int grp = lane >> 4, l16 = lane & 15;

  const int wm = m0 + wave * 32;                // wave's 32-row window
  const __bf16* a0 = hs + (size_t)(wm + l16) * HDIM;
  const __bf16* a1 = hs + (size_t)(wm + 16 + l16) * HDIM;

  f32x8 acc00 = {}, acc01 = {}, acc10 = {}, acc11 = {};

  for (int pass = 0; pass < HDIM / KCHUNK; ++pass) {
    const int kb = pass * KCHUNK;

    // Stage W[n0:n0+32, kb:kb+512] into LDS: 2048 x 16B chunks, 8 per thread.
#pragma unroll
    for (int j = 0; j < 8; ++j) {
      int chunk = tid + 256 * j;                // 0..2047
      int row   = chunk >> 6;                   // 64 chunks per row
      int koff  = (chunk & 63) * 8;             // halves within chunk row
      int rowg  = n0 + row;
      if (rowg > VDIM - 1) rowg = VDIM - 1;     // clamp (partial last block)
      async_cp16(&lds_w[row * LDSROW + koff],
                 Wout + (size_t)rowg * HDIM + kb + koff);
    }
    wait_async();
    __syncthreads();

    const __bf16* b0 = &lds_w[(size_t)l16 * LDSROW];
    const __bf16* b1 = &lds_w[(size_t)(16 + l16) * LDSROW];
#pragma unroll 2
    for (int k0 = 0; k0 < KCHUNK; k0 += 32) {
      bf16x16 fa0 = load_a_frag(a0, kb + k0, grp);
      bf16x16 fa1 = load_a_frag(a1, kb + k0, grp);
      bf16x16 fb0 = load_b_frag(b0, k0, grp);
      bf16x16 fb1 = load_b_frag(b1, k0, grp);
      acc00 = wmma_bf16(fa0, fb0, acc00);
      acc01 = wmma_bf16(fa0, fb1, acc01);
      acc10 = wmma_bf16(fa1, fb0, acc10);
      acc11 = wmma_bf16(fa1, fb1, acc11);
    }
    __syncthreads();  // protect LDS before next stage overwrites
  }

  // Epilogue: + b_out, guarded stores (only the last n-block is partial).
  int col0 = n0 + l16;
  int col1 = n0 + 16 + l16;
  float bn0 = (col0 < VDIM) ? bout[col0] : 0.0f;
  float bn1 = (col1 < VDIM) ? bout[col1] : 0.0f;
  const f32x8* accs[2][2] = {{&acc00, &acc01}, {&acc10, &acc11}};
#pragma unroll
  for (int mm = 0; mm < 2; ++mm) {
#pragma unroll
    for (int r = 0; r < 8; ++r) {
      size_t row = (size_t)(wm + mm * 16 + grp * 8 + r);
      if (col0 < VDIM) logits[row * VDIM + col0] = (*accs[mm][0])[r] + bn0;
      if (col1 < VDIM) logits[row * VDIM + col1] = (*accs[mm][1])[r] + bn1;
    }
  }
}

// ---------------------------------------------------------------------------
// Host launcher
// ---------------------------------------------------------------------------
extern "C" void kernel_launch(void* const* d_in, const int* in_sizes, int n_in,
                              void* d_out, int out_size, void* d_ws, size_t ws_size,
                              hipStream_t stream) {
  (void)in_sizes; (void)n_in; (void)out_size; (void)ws_size;

  const int*   inputs = (const int*)  d_in[0];  // [T, B] int32
  const float* h0     = (const float*)d_in[1];  // [B, H]
  const float* c0     = (const float*)d_in[2];  // [B, H]
  const float* emb    = (const float*)d_in[3];  // [V, D]
  const float* W_ih   = (const float*)d_in[4];  // [4H, D]
  const float* W_hh   = (const float*)d_in[5];  // [4H, H]
  const float* bias   = (const float*)d_in[6];  // [4H]
  const float* W_out  = (const float*)d_in[7];  // [V, H]
  const float* b_out  = (const float*)d_in[8];  // [V]

  float* out    = (float*)d_out;
  float* logits = out;                                      // [T*B, V]
  float* hT     = out + (size_t)TDIM * BDIM * VDIM;         // [B, H]
  float* cT     = hT + (size_t)BDIM * HDIM;                 // [B, H]

  // Workspace carve-up (256B aligned slabs).
  char* base = (char*)d_ws;
  size_t off = 0;
  auto carve = [&](size_t bytes) -> char* {
    char* p = base + off;
    off += (bytes + 255) & ~(size_t)255;
    return p;
  };
  __bf16* Xbf    = (__bf16*)carve((size_t)TDIM * BDIM * DDIM * 2); // [T*B, D]
  __bf16* Wih_bf = (__bf16*)carve((size_t)4 * HDIM * DDIM * 2);    // [4H, D]
  __bf16* Whh_bf = (__bf16*)carve((size_t)4 * HDIM * HDIM * 2);    // [4H, H]
  __bf16* Wout_bf= (__bf16*)carve((size_t)VDIM * HDIM * 2);        // [V, H]
  __bf16* h0_bf  = (__bf16*)carve((size_t)BDIM * HDIM * 2);        // [B, H]
  __bf16* hs_bf  = (__bf16*)carve((size_t)TDIM * BDIM * HDIM * 2); // [T*B, H]
  float*  gates  = (float*) carve((size_t)BDIM * 4 * HDIM * 4);    // [B, 4H]

  const int TPB = 256;
  auto blocks = [](size_t n, int tpb) { return (unsigned)((n + tpb - 1) / tpb); };

  // One-time (per call) conversions + gather.
  cast_bf16_kernel<<<blocks((size_t)4 * HDIM * DDIM, TPB), TPB, 0, stream>>>(
      W_ih, Wih_bf, 4 * HDIM * DDIM);
  cast_bf16_kernel<<<blocks((size_t)4 * HDIM * HDIM, TPB), TPB, 0, stream>>>(
      W_hh, Whh_bf, 4 * HDIM * HDIM);
  cast_bf16_kernel<<<blocks((size_t)VDIM * HDIM, TPB), TPB, 0, stream>>>(
      W_out, Wout_bf, VDIM * HDIM);
  embed_kernel<<<blocks((size_t)TDIM * BDIM * DDIM, TPB), TPB, 0, stream>>>(
      inputs, emb, Xbf);
  init_state_kernel<<<blocks((size_t)BDIM * HDIM, TPB), TPB, 0, stream>>>(
      h0, c0, hT, cT, h0_bf);

  // Sequential scan: 2 kernels per timestep.
  const unsigned gate_blocks =
      ((BDIM / 32) * ((4 * HDIM) / 32)) / (TPB / 32);       // 32
  const unsigned cell_blocks = blocks((size_t)BDIM * HDIM, TPB);
  for (int t = 0; t < TDIM; ++t) {
    const __bf16* Xt    = Xbf + (size_t)t * BDIM * DDIM;
    const __bf16* hprev = (t == 0) ? h0_bf : (hs_bf + (size_t)(t - 1) * BDIM * HDIM);
    lstm_gates_kernel<<<gate_blocks, TPB, 0, stream>>>(
        Xt, hprev, Wih_bf, Whh_bf, bias, gates);
    lstm_cell_kernel<<<cell_blocks, TPB, 0, stream>>>(
        gates, hT, cT, hs_bf + (size_t)t * BDIM * HDIM);
  }

  // Output projection over all timesteps at once.
  const unsigned proj_blocks = (TDIM * BDIM / PROJ_BM) * NBLK;  // 64*313
  proj_kernel<<<proj_blocks, TPB, 0, stream>>>(hs_bf, Wout_bf, b_out, logits);
}
